// MDFM_3590592660121
// MI455X (gfx1250) — matrix operand order repaired and checked by reading, not compiled, not run on hardware
//
#include <hip/hip_runtime.h>
#include <hip/hip_bf16.h>

typedef __attribute__((ext_vector_type(16))) _Float16 v16h;
typedef __attribute__((ext_vector_type(8)))  float    v8f;

#define HH 96
#define WW 96
#define HW 9216
#define CIN 128
#define BB 8
#define GRP 8
#define KB 4
#define COUT 256

__device__ __forceinline__ int refl(int p, int n) {
  p = (p < 0) ? -p : p;
  p = (p >= n) ? (2 * n - 2 - p) : p;
  return p;
}
__device__ __forceinline__ float gelu_exact(float x) {
  return 0.5f * x * (1.0f + erff(x * 0.70710678118654752f));
}

// ---------------------------------------------------------------------------
// Kernel 1: per (b,c) channel image -> 9 reflect-shifted spatial means.
// meanP[b*128+c][j], j = di*3+dj (patch ordering of the reference).
// meanP[...][4] is the plain channel mean (used for the exact FFT collapse).
// ---------------------------------------------------------------------------
__global__ __launch_bounds__(256) void k_meanpatch(const float* __restrict__ x,
                                                   float* __restrict__ meanP) {
  __shared__ float img[HW];          // 36 KB: whole 96x96 channel in LDS
  __shared__ float red[9];
  const int bc = blockIdx.x;         // 0..1023
  const float* src = x + (size_t)bc * HW;
  for (int t = threadIdx.x; t < HW; t += 256) img[t] = src[t];
  if (threadIdx.x < 9) red[threadIdx.x] = 0.f;
  __syncthreads();
  float part[9];
#pragma unroll
  for (int j = 0; j < 9; j++) part[j] = 0.f;
  for (int p = threadIdx.x; p < HW; p += 256) {
    const int y = p / WW, xx = p % WW;
#pragma unroll
    for (int di = 0; di < 3; di++) {
      const int yy = refl(y + di - 1, HH) * WW;
#pragma unroll
      for (int dj = 0; dj < 3; dj++)
        part[di * 3 + dj] += img[yy + refl(xx + dj - 1, WW)];
    }
  }
#pragma unroll
  for (int j = 0; j < 9; j++) atomicAdd(&red[j], part[j]);
  __syncthreads();
  if (threadIdx.x < 9) meanP[bc * 9 + threadIdx.x] = red[threadIdx.x] * (1.0f / HW);
}

// ---------------------------------------------------------------------------
// Kernel 2 (tiny, one block): exact algebraic collapse of the FFT/band/BN/
// softmax/SE path, producing:
//   Wcomb[b][d][i][j] = w_mlp1[d,i] * sigmoid_gate[b,flat] * filt[b,g,k,j]
//   w2h: w_mlp2 repacked as f16 in WMMA A-fragment order.
// ---------------------------------------------------------------------------
__global__ __launch_bounds__(512) void k_small(
    const float* __restrict__ meanP, const float* __restrict__ w1,
    const float* __restrict__ w2, const float* __restrict__ bng,
    const float* __restrict__ bnb, const float* __restrict__ bnm,
    const float* __restrict__ bnv, const float* __restrict__ wsca1,
    const float* __restrict__ bsca1, const float* __restrict__ wsca2,
    const float* __restrict__ bsca2, const float* __restrict__ wmlp1,
    const float* __restrict__ wmlp2, float* __restrict__ Wcomb,
    _Float16* __restrict__ w2h) {
  __shared__ float sfilt[BB * GRP * KB * 9];  // 2304
  __shared__ float smy[BB * 512];             // 4096
  __shared__ float ss1[BB * 128];             // 1024
  __shared__ float ssg[BB * 512];             // 4096
  const int tid = threadIdx.x;

  // A) df -> BN -> softmax. FFT collapse: df[b,g,k] = mean(x1[b,g]) * [k==0].
  for (int t = tid; t < BB * GRP * KB; t += blockDim.x) {
    const int b = t >> 5, gk = t & 31, g = gk >> 2, k = gk & 3;
    float df0 = 0.f;
    if (k == 0)
      for (int c = 0; c < 16; c++)
        df0 += w1[g * 16 + c] * meanP[(b * CIN + g * 16 + c) * 9 + 4];
    float v[9], mx = -1e30f;
    for (int j = 0; j < 9; j++) {
      const int idx = gk * 9 + j;
      float d = df0 * w2[idx];
      d = (d - bnm[idx]) * rsqrtf(bnv[idx] + 1e-5f) * bng[idx] + bnb[idx];
      v[j] = d; mx = fmaxf(mx, d);
    }
    float s = 0.f;
    for (int j = 0; j < 9; j++) { v[j] = __expf(v[j] - mx); s += v[j]; }
    const float inv = 1.f / s;
    for (int j = 0; j < 9; j++) sfilt[t * 9 + j] = v[j] * inv;
  }
  __syncthreads();
  // B) mean_y[b,flat] = sum_j filt * meanPatch  (exact spatial mean of y)
  for (int t = tid; t < BB * 512; t += blockDim.x) {
    const int b = t >> 9, flat = t & 511;
    const int g = flat >> 6, k = (flat >> 4) & 3, c16 = flat & 15;
    const float* f = &sfilt[((b * GRP + g) * KB + k) * 9];
    const float* mp = &meanP[(b * CIN + g * 16 + c16) * 9];
    float a = 0.f;
    for (int j = 0; j < 9; j++) a += f[j] * mp[j];
    smy[t] = a;
  }
  __syncthreads();
  // C) SE hidden: gelu(mean_y @ w_sca1^T + b)
  for (int t = tid; t < BB * 128; t += blockDim.x) {
    const int b = t >> 7, o = t & 127;
    const float* row = &wsca1[o * 512];
    const float* mv = &smy[b * 512];
    float a = bsca1[o];
    for (int i = 0; i < 512; i++) a += mv[i] * row[i];
    ss1[t] = gelu_exact(a);
  }
  __syncthreads();
  // D) SE gate: sigmoid(s1 @ w_sca2^T + b)
  for (int t = tid; t < BB * 512; t += blockDim.x) {
    const int b = t >> 9, o = t & 511;
    const float* row = &wsca2[o * 128];
    const float* sv = &ss1[b * 128];
    float a = bsca2[o];
    for (int i = 0; i < 128; i++) a += sv[i] * row[i];
    ssg[t] = 1.f / (1.f + __expf(-a));
  }
  __syncthreads();
  // E) fold gate+filt+w_mlp1 into one 36-tap stencil weight per (b,d)
  for (int t = tid; t < BB * CIN * 4 * 9; t += blockDim.x) {
    const int j = t % 9;
    int r = t / 9;
    const int i = r & 3; r >>= 2;
    const int d = r & 127, b = r >> 7;
    const int g = d >> 4, k = (d >> 2) & 3;
    const int flat = g * 64 + k * 16 + (d & 3) * 4 + i;
    Wcomb[t] = wmlp1[d * 4 + i] * ssg[b * 512 + flat] *
               sfilt[((b * GRP + g) * KB + k) * 9 + j];
  }
  // F) pack w_mlp2 -> f16 as WMMA A fragments, contiguous k order (same as B)
  //    lane L: row = L&15, k(e) = (L>>4)*16 + e
  for (int t = tid; t < 16 * 4 * 32 * 16; t += blockDim.x) {
    const int e = t & 15, L = (t >> 4) & 31, C = (t >> 9) & 3, T = t >> 11;
    const int o = T * 16 + (L & 15);
    const int d = C * 32 + (L >> 4) * 16 + e;
    w2h[t] = (_Float16)wmlp2[o * 128 + d];
  }
}

// ---------------------------------------------------------------------------
// Kernel 3: fused main pass. Each wave = 16 consecutive pixels of one row.
// Lane computes 16 m_d values (36-tap stencil + GELU) forming the WMMA B
// fragment directly; 16 o-tiles x 4 k-chunks of v_wmma_f32_16x16x32_f16
// produce out[256] per pixel, coalesced stores with bias.
// o-tile loop is unrolled only 4x to keep the live set (bfrag + 4 acc +
// 4 A-fragments) under 256 VGPRs -> no s_set_vgpr_msb churn, >=4 waves/SIMD.
// ---------------------------------------------------------------------------
__global__ __launch_bounds__(128) void k_main(
    const float* __restrict__ x, const float* __restrict__ Wcomb,
    const _Float16* __restrict__ w2h, const float* __restrict__ bmlp1,
    const float* __restrict__ bmlp2, float* __restrict__ out) {
  __shared__ __align__(16) float sWc[CIN * 36];  // 18 KB per-batch stencil weights
  __shared__ float sB1[128];
  __shared__ float sB2[256];
  const int b = blockIdx.x / 144;
  const int blk = blockIdx.x % 144;
  {
    const float* wcb = Wcomb + (size_t)b * (CIN * 36);
    for (int t = threadIdx.x; t < CIN * 36; t += 128) sWc[t] = wcb[t];
    if (threadIdx.x < 128) sB1[threadIdx.x] = bmlp1[threadIdx.x];
    for (int t = threadIdx.x; t < 256; t += 128) sB2[t] = bmlp2[t];
  }
  __syncthreads();
  const int wave = threadIdx.x >> 5;
  const int lane = threadIdx.x & 31;
  const int gb = blk * 4 + wave;  // 0..575 group-in-batch
  const int y = gb / 6;
  const int x0 = (gb % 6) * 16;
  const int half = lane >> 4;
  const int px = x0 + (lane & 15);

  v16h bfrag[4];
#pragma unroll
  for (int C = 0; C < 4; C++) {
    const int G = C * 2 + half;          // channel group this lane-half covers
    const int D0 = C * 32 + half * 16;   // base d of this lane's 16 m-values
    const float* xb = x + ((size_t)b * CIN + G * 16) * HW;
    __builtin_prefetch(xb + (size_t)HW * 16, 0, 0);
    float m[16];
#pragma unroll
    for (int t = 0; t < 16; t++) m[t] = 0.f;
#pragma unroll
    for (int c16 = 0; c16 < 16; c16++) {
      const float* xc = xb + c16 * HW;
      float nbr[9];
#pragma unroll
      for (int di = 0; di < 3; di++) {
        const int yy = refl(y + di - 1, HH) * WW;
#pragma unroll
        for (int dj = 0; dj < 3; dj++)
          nbr[di * 3 + dj] = xc[yy + refl(px + dj - 1, WW)];
      }
      const int dlow = c16 >> 2, ii = c16 & 3;
#pragma unroll
      for (int k = 0; k < 4; k++) {
        const int d = D0 + k * 4 + dlow;
        const float* Wc = &sWc[(d * 4 + ii) * 9];
        float a = 0.f;
#pragma unroll
        for (int j = 0; j < 9; j++) a = fmaf(nbr[j], Wc[j], a);
        m[k * 4 + dlow] += a;
      }
    }
    v16h bf;
#pragma unroll
    for (int t = 0; t < 16; t++)
      bf[t] = (_Float16)gelu_exact(m[t] + sB1[D0 + t]);
    bfrag[C] = bf;
  }

  const v16h* aW = (const v16h*)w2h;  // A fragments, L2-resident (64 KB)
#pragma unroll 4
  for (int T = 0; T < 16; T++) {
    v8f acc;
#pragma unroll
    for (int r = 0; r < 8; r++) acc[r] = 0.f;
#pragma unroll
    for (int C = 0; C < 4; C++) {
      v16h a = aW[(T * 4 + C) * 32 + lane];
      acc = __builtin_amdgcn_wmma_f32_16x16x32_f16(false, a, false, bfrag[C],
                                                   (short)0, acc, false, false);
    }
#pragma unroll
    for (int r = 0; r < 8; r++) {
      const int o = T * 16 + half * 8 + r;
      out[((size_t)(b * COUT + o)) * HW + y * WW + px] = acc[r] + sB2[o];
    }
  }
}

extern "C" void kernel_launch(void* const* d_in, const int* in_sizes, int n_in,
                              void* d_out, int out_size, void* d_ws,
                              size_t ws_size, hipStream_t stream) {
  (void)in_sizes; (void)n_in; (void)out_size; (void)ws_size;
  const float* x_f   = (const float*)d_in[0];
  const float* w1    = (const float*)d_in[1];
  const float* w2    = (const float*)d_in[2];
  const float* bng   = (const float*)d_in[3];
  const float* bnb   = (const float*)d_in[4];
  const float* bnm   = (const float*)d_in[5];
  const float* bnv   = (const float*)d_in[6];
  const float* wsca1 = (const float*)d_in[7];
  const float* bsca1 = (const float*)d_in[8];
  const float* wsca2 = (const float*)d_in[9];
  const float* bsca2 = (const float*)d_in[10];
  const float* wmlp1 = (const float*)d_in[11];
  const float* bmlp1 = (const float*)d_in[12];
  const float* wmlp2 = (const float*)d_in[13];
  const float* bmlp2 = (const float*)d_in[14];

  char* ws = (char*)d_ws;
  float*    meanP = (float*)ws;                         // 9216 floats
  float*    Wcomb = (float*)(ws + 36864);               // 36864 floats
  _Float16* w2h   = (_Float16*)(ws + 36864 + 147456);   // 32768 halfs

  k_meanpatch<<<BB * CIN, 256, 0, stream>>>(x_f, meanP);
  k_small<<<1, 512, 0, stream>>>(meanP, w1, w2, bng, bnb, bnm, bnv, wsca1,
                                 bsca1, wsca2, bsca2, wmlp1, wmlp2, Wcomb, w2h);
  k_main<<<BB * 144, 128, 0, stream>>>(x_f, Wcomb, w2h, bmlp1, bmlp2,
                                       (float*)d_out);
}
